// CausalConv3dAttnBlock_7567732376026
// MI455X (gfx1250) — matrix operand order repaired and checked by reading, not compiled, not run on hardware
//
#include <hip/hip_runtime.h>
#include <hip/hip_bf16.h>

typedef __attribute__((ext_vector_type(8)))  _Float16 v8h;
typedef __attribute__((ext_vector_type(16))) _Float16 v16h;
typedef __attribute__((ext_vector_type(8)))  float    v8f;

constexpr int CH  = 512;      // channels
constexpr int TFR = 4;        // frames
constexpr int SP  = 4096;     // h*w tokens per frame
constexpr int CSP = TFR * SP; // 16384 elements per channel
constexpr int GSZ = 16 * CSP; // 262144 elements per group (16 ch/group)
constexpr float QSCALE = 0.04419417382415922f; // 512^-0.5

// ---- WMMA operand loaders (layouts per CDNA5 ISA 7.12.2) ----
// A 16x32 f16: lane = (m & 15) + 16*half; elems 0..7 <- K = half*8 + 0..7,
//              elems 8..15 <- K = half*8 + 16..23.
// `row` points at A[m][k0] in a row-major matrix.
__device__ __forceinline__ v16h load_a32(const _Float16* row) {
  int kh = ((threadIdx.x >> 4) & 1) << 3;
  v8h lo = *(const v8h*)(row + kh);
  v8h hi = *(const v8h*)(row + kh + 16);
  return __builtin_shufflevector(lo, hi, 0,1,2,3,4,5,6,7,8,9,10,11,12,13,14,15);
}
// B 32x16 f16 given as B^T rows: lane n = lane&15; elems 0..7 <- K = half*8 + 0..7, elems 8..15 <- K = half*8 + 16..23.
// `row` points at B^T[n][k0] (i.e. 32 contiguous K values of column n).
__device__ __forceinline__ v16h load_b32(const _Float16* row) {
  int kh = ((threadIdx.x >> 4) & 1) << 3;
  v8h lo = *(const v8h*)(row + kh);
  v8h hi = *(const v8h*)(row + kh + 16);
  return __builtin_shufflevector(lo, hi, 0,1,2,3,4,5,6,7,8,9,10,11,12,13,14,15);
}

#define WMMA_F16(a, b, c) __builtin_amdgcn_wmma_f32_16x16x32_f16(false, (a), false, (b), (short)0, (c), false, false)

// ---------------- K1: group statistics ----------------
__global__ void gn_stats(const float* __restrict__ x, float* __restrict__ stats) {
  int g = blockIdx.x;
  const float4* p = (const float4*)(x + (size_t)g * GSZ);
  int tid = threadIdx.x;
  float s = 0.f, s2 = 0.f;
  for (int i = tid; i < GSZ / 4; i += 256) {
    float4 v = p[i];
    s  += v.x + v.y + v.z + v.w;
    s2 += v.x*v.x + v.y*v.y + v.z*v.z + v.w*v.w;
  }
  __shared__ float rs[256], rs2[256];
  rs[tid] = s; rs2[tid] = s2;
  __syncthreads();
  for (int st = 128; st > 0; st >>= 1) {
    if (tid < st) { rs[tid] += rs[tid + st]; rs2[tid] += rs2[tid + st]; }
    __syncthreads();
  }
  if (tid == 0) {
    float n = (float)GSZ;
    float mean = rs[0] / n;
    float var  = rs2[0] / n - mean * mean;
    stats[2*g]   = mean;
    stats[2*g+1] = rsqrtf(var + 1e-6f);
  }
}

// ---------------- K2: weights fp32 -> f16 (q,k,v,p contiguous) ----------------
__global__ void w_to_f16(const float* __restrict__ wq, const float* __restrict__ wk,
                         const float* __restrict__ wv, const float* __restrict__ wp,
                         _Float16* __restrict__ dst) {
  size_t gid = (size_t)blockIdx.x * 256 + threadIdx.x;   // < 4*CH*CH exactly
  int m = (int)(gid >> 18);
  int idx = (int)(gid & ((1u << 18) - 1));
  const float* s = (m == 0) ? wq : (m == 1) ? wk : (m == 2) ? wv : wp;
  dst[gid] = (_Float16)s[idx];
}

// ---------------- K3: normalize + transpose to (t, s, c) f16 tokens ----------------
__global__ void gn_normalize_pack(const float* __restrict__ x,
                                  const float* __restrict__ gamma,
                                  const float* __restrict__ beta,
                                  const float* __restrict__ stats,
                                  _Float16* __restrict__ tok) {
  __shared__ _Float16 tile[32][65];  // padded to spread LDS banks
  int bid = blockIdx.x;
  int cb = bid & 15;            // 16 channel blocks of 32
  int sb = (bid >> 4) & 63;     // 64 spatial blocks of 64
  int t  = bid >> 10;           // 4 frames
  int c0 = cb * 32, s0 = sb * 64;
  int tid = threadIdx.x;
  for (int i = tid; i < 32 * 64; i += 256) {
    int cl = i >> 6, sl = i & 63;
    int c = c0 + cl;
    int g = c >> 4;
    float mean = stats[2*g], rstd = stats[2*g+1];
    float v = x[(size_t)c * CSP + (size_t)t * SP + s0 + sl];
    v = (v - mean) * rstd * gamma[c] + beta[c];
    tile[cl][sl] = (_Float16)v;
  }
  __syncthreads();
  for (int i = tid; i < 32 * 64; i += 256) {
    int sl = i >> 5, cl = i & 31;
    tok[((size_t)t * SP + s0 + sl) * CH + c0 + cl] = tile[cl][sl];
  }
}

// ---------------- K4: QKV projection (WMMA), q scaled, v stored transposed ----------------
__global__ void qkv_proj(const _Float16* __restrict__ tok,
                         const _Float16* __restrict__ w16,
                         const float* __restrict__ bq, const float* __restrict__ bk,
                         const float* __restrict__ bv,
                         _Float16* __restrict__ q, _Float16* __restrict__ k,
                         _Float16* __restrict__ vT) {
  int tid = threadIdx.x;
  int wave = tid >> 5, lane = tid & 31;
  int id = blockIdx.x * 8 + wave;            // < 3*4*256*32 = 98304
  int nt = id & 31;
  int mt = (id >> 5) & 255;
  int t  = (id >> 13) & 3;
  int p  = id >> 15;                          // 0=q 1=k 2=v
  int row = lane & 15;
  const _Float16* W   = w16 + (size_t)p * CH * CH;
  const float* bias   = (p == 0) ? bq : (p == 1) ? bk : bv;
  const _Float16* arow = tok + ((size_t)t * SP + mt * 16 + row) * CH;
  const _Float16* brow = W + (size_t)(nt * 16 + row) * CH;
  v8f acc = {};
  for (int kk = 0; kk < CH; kk += 32)
    acc = WMMA_F16(load_a32(arow + kk), load_b32(brow + kk), acc);
  int n = nt * 16 + row;                      // D col = lane & 15
  float bn = bias[n];
  int rbase = (lane >> 4) << 3;               // D row = rbase + r
  if (p == 2) {
    _Float16* dst = vT + (size_t)t * CH * SP;
#pragma unroll
    for (int r = 0; r < 8; ++r)
      dst[(size_t)n * SP + mt * 16 + rbase + r] = (_Float16)(acc[r] + bn);
  } else {
    _Float16* dst = ((p == 0) ? q : k) + ((size_t)t * SP + mt * 16) * CH;
    float sc = (p == 0) ? QSCALE : 1.0f;
#pragma unroll
    for (int r = 0; r < 8; ++r)
      dst[(size_t)(rbase + r) * CH + n] = (_Float16)((acc[r] + bn) * sc);
  }
}

// ---------------- K5: flash attention (per-frame, 16 q rows / block) ----------------
__global__ void attn_flash(const _Float16* __restrict__ q,
                           const _Float16* __restrict__ k,
                           const _Float16* __restrict__ vT,
                           _Float16* __restrict__ ao) {
  constexpr int KVC = 128;
  __shared__ __align__(16) _Float16 qs[16 * CH];   // 16 KB
  __shared__ float    S[16 * KVC];                 // 8 KB scores
  __shared__ __align__(16) _Float16 P[16 * KVC];   // 4 KB probs
  __shared__ float pm[16 * 8], ps[16 * 8];
  __shared__ float mrun[16], lrun[16], corr[16], mnew[16];

  int blk = blockIdx.x;
  int t  = blk >> 8;
  int qb = blk & 255;
  int tid = threadIdx.x;
  int wave = tid >> 5, lane = tid & 31;
  int row16 = lane & 15;
  int rbase = (lane >> 4) << 3;

  const _Float16* qbase = q  + ((size_t)t * SP + qb * 16) * CH;
  const _Float16* kbase = k  + (size_t)t * SP * CH;
  const _Float16* vbase = vT + (size_t)t * CH * SP;

  // Stage q tile into LDS with CDNA5 async memory->LDS copies (ASYNCcnt path).
  // Low 32 bits of a generic shared pointer are the wave-relative LDS offset
  // (flat->LDS mapping, ISA 10.2): feed them as the per-lane dsaddr VGPR.
  {
    unsigned lds0 = (unsigned)(uintptr_t)(&qs[0]);
    for (int i = tid; i < 16 * CH / 8; i += 256) {
      unsigned dsa = lds0 + (unsigned)i * 16u;        // 16B per B128 copy
      const _Float16* g = qbase + (size_t)i * 8;
      asm volatile("global_load_async_to_lds_b128 %0, %1, off"
                   :: "v"(dsa), "v"(g) : "memory");
    }
    asm volatile("s_wait_asynccnt 0" ::: "memory");
  }
  if (tid < 16) { mrun[tid] = -1e30f; lrun[tid] = 0.f; }
  __syncthreads();

  v8f acc[4] = {};                               // 16 rows x 64 cols (wave's C slice)
  const _Float16* qrow = qs + row16 * CH;

  for (int kv0 = 0; kv0 < SP; kv0 += KVC) {
    // Prefetch next chunk's K row and V rows for this wave (global_prefetch_b8).
    if (kv0 + KVC < SP) {
      __builtin_prefetch(kbase + (size_t)(kv0 + KVC + wave * 16 + row16) * CH, 0, 1);
      __builtin_prefetch(vbase + (size_t)(wave * 64 + row16) * SP + kv0 + KVC, 0, 1);
    }
    // --- S tile: wave w covers kv cols [kv0+16w, kv0+16w+16) ---
    v8f s = {};
    const _Float16* krow = kbase + (size_t)(kv0 + wave * 16 + row16) * CH;
    for (int kk = 0; kk < CH; kk += 32)
      s = WMMA_F16(load_a32(qrow + kk), load_b32(krow + kk), s);
    {
      int col = wave * 16 + (lane & 15);
#pragma unroll
      for (int r = 0; r < 8; ++r) S[(rbase + r) * KVC + col] = s[r];
    }
    __syncthreads();
    // --- online softmax: partial row max ---
    if (tid < 128) {
      int row = tid >> 3, seg = tid & 7;
      const float* sr = &S[row * KVC + seg * 16];
      float m = sr[0];
      for (int j = 1; j < 16; ++j) m = fmaxf(m, sr[j]);
      pm[row * 8 + seg] = m;
    }
    __syncthreads();
    if (tid < 16) {
      float m = mrun[tid];
      for (int j = 0; j < 8; ++j) m = fmaxf(m, pm[tid * 8 + j]);
      mnew[tid] = m;
      corr[tid] = __expf(mrun[tid] - m);
      mrun[tid] = m;
    }
    __syncthreads();
    if (tid < 128) {
      int row = tid >> 3, seg = tid & 7;
      float m = mnew[row];
      float sum = 0.f;
      const float* sr = &S[row * KVC + seg * 16];
      _Float16* pr = &P[row * KVC + seg * 16];
      for (int j = 0; j < 16; ++j) {
        float pv = __expf(sr[j] - m);
        pr[j] = (_Float16)pv;
        sum += pv;
      }
      ps[row * 8 + seg] = sum;
    }
    __syncthreads();
    if (tid < 16) {
      float l = lrun[tid] * corr[tid];
      for (int j = 0; j < 8; ++j) l += ps[tid * 8 + j];
      lrun[tid] = l;
    }
    // --- rescale O, then O += P @ V (wave owns C cols [64w, 64w+64)) ---
#pragma unroll
    for (int r = 0; r < 8; ++r) {
      float cf = corr[rbase + r];
#pragma unroll
      for (int n2 = 0; n2 < 4; ++n2) acc[n2][r] *= cf;
    }
    const _Float16* prow = P + row16 * KVC;
#pragma unroll
    for (int n2 = 0; n2 < 4; ++n2) {
      const _Float16* vrow = vbase + (size_t)(wave * 64 + n2 * 16 + row16) * SP + kv0;
#pragma unroll
      for (int ks = 0; ks < KVC; ks += 32)
        acc[n2] = WMMA_F16(load_a32(prow + ks), load_b32(vrow + ks), acc[n2]);
    }
    __syncthreads();
  }
  // --- finalize: divide by l, store f16 attention output (t, s, c) ---
  _Float16* obase = ao + ((size_t)t * SP + qb * 16) * CH;
  int col = lane & 15;
#pragma unroll
  for (int r = 0; r < 8; ++r) {
    float inv = 1.0f / lrun[rbase + r];
#pragma unroll
    for (int n2 = 0; n2 < 4; ++n2)
      obase[(size_t)(rbase + r) * CH + wave * 64 + n2 * 16 + col] =
          (_Float16)(acc[n2][r] * inv);
  }
}

// ---------------- K6: output projection + bias + residual ----------------
__global__ void out_proj(const _Float16* __restrict__ ao,
                         const _Float16* __restrict__ wp16,
                         const float* __restrict__ bp,
                         const float* __restrict__ x,
                         float* __restrict__ out) {
  int tid = threadIdx.x;
  int wave = tid >> 5, lane = tid & 31;
  int id = blockIdx.x * 8 + wave;               // < 4*256*32 = 32768
  int nt = id & 31;
  int mt = (id >> 5) & 255;
  int t  = id >> 13;
  int row = lane & 15;
  const _Float16* arow = ao  + ((size_t)t * SP + mt * 16 + row) * CH;
  const _Float16* brow = wp16 + (size_t)(nt * 16 + row) * CH;
  v8f acc = {};
  for (int kk = 0; kk < CH; kk += 32)
    acc = WMMA_F16(load_a32(arow + kk), load_b32(brow + kk), acc);
  int n = nt * 16 + row;
  float bn = bp[n];
  int rbase = (lane >> 4) << 3;
  size_t obase = (size_t)n * CSP + (size_t)t * SP + mt * 16 + rbase;
#pragma unroll
  for (int r = 0; r < 8; ++r)
    out[obase + r] = x[obase + r] + acc[r] + bn;
}

// ---------------- launch ----------------
extern "C" void kernel_launch(void* const* d_in, const int* in_sizes, int n_in,
                              void* d_out, int out_size, void* d_ws, size_t ws_size,
                              hipStream_t stream) {
  const float* x     = (const float*)d_in[0];
  const float* gamma = (const float*)d_in[1];
  const float* beta  = (const float*)d_in[2];
  const float* wq    = (const float*)d_in[3];
  const float* bq    = (const float*)d_in[4];
  const float* wk    = (const float*)d_in[5];
  const float* bk    = (const float*)d_in[6];
  const float* wv    = (const float*)d_in[7];
  const float* bv    = (const float*)d_in[8];
  const float* wp    = (const float*)d_in[9];
  const float* bp    = (const float*)d_in[10];
  float* out = (float*)d_out;

  char* ws = (char*)d_ws;
  size_t off = 0;
  float* stats = (float*)(ws + off);           off += 256;
  const size_t tokB = (size_t)TFR * SP * CH * sizeof(_Float16);  // 16 MB
  _Float16* tok = (_Float16*)(ws + off);       off += tokB;
  _Float16* w16 = (_Float16*)(ws + off);       off += 4ull * CH * CH * sizeof(_Float16);
  _Float16* q16 = (_Float16*)(ws + off);       off += tokB;
  _Float16* k16 = (_Float16*)(ws + off);       off += tokB;
  _Float16* vT  = (_Float16*)(ws + off);       off += tokB;
  _Float16* ao  = (_Float16*)(ws + off);       off += tokB;
  (void)ws_size; (void)n_in; (void)in_sizes; (void)out_size;

  gn_stats<<<32, 256, 0, stream>>>(x, stats);
  w_to_f16<<<(4 * CH * CH) / 256, 256, 0, stream>>>(wq, wk, wv, wp, w16);
  gn_normalize_pack<<<16 * 64 * TFR, 256, 0, stream>>>(x, gamma, beta, stats, tok);
  qkv_proj<<<(3 * TFR * 256 * 32) / 8, 256, 0, stream>>>(tok, w16, bq, bk, bv, q16, k16, vT);
  attn_flash<<<TFR * 256, 256, 0, stream>>>(q16, k16, vT, ao);
  out_proj<<<(TFR * 256 * 32) / 8, 256, 0, stream>>>(ao, w16 + 3 * CH * CH, bp, x, out);
}
